// Head_30081950941627
// MI455X (gfx1250) — matrix operand order repaired and checked by reading, not compiled, not run on hardware
//
#include <hip/hip_runtime.h>

typedef __attribute__((ext_vector_type(16))) __bf16 v16bf;
typedef __attribute__((ext_vector_type(8)))  __bf16 v8bf;
typedef __attribute__((ext_vector_type(8)))  float  v8f;

#define BATCH 8
#define SLEN  4096
#define DM    512
#define DKV   64

__device__ __forceinline__ v16bf load_frag(const __bf16* p0, const __bf16* p1) {
  v8bf lo = *(const v8bf*)p0;
  v8bf hi = *(const v8bf*)p1;
  v16bf r;
#pragma unroll
  for (int i = 0; i < 8; ++i) { r[i] = lo[i]; r[i + 8] = hi[i]; }
  return r;
}

__device__ __forceinline__ v8f wmma_bf16(v16bf a, v16bf b, v8f c) {
  // D = A(16x32 bf16) * B(32x16 bf16) + C(16x16 f32)
  return __builtin_amdgcn_wmma_f32_16x16x32_bf16(false, a, false, b, (short)0, c,
                                                 false, false);
}

// Async copy of a 64-row x 64-bf16 tile (row stride 64 elems in global) into a
// padded LDS tile (row stride 72 elems).  512 x 16B chunks spread over 128
// threads; each lane issues its own global_load_async_to_lds_b128 (ASYNCcnt).
// LDS byte address = low 32 bits of the generic shared pointer (aperture map).
__device__ __forceinline__ void async_tile_64x64(const __bf16* g, __bf16* l,
                                                 int tid) {
#pragma unroll
  for (int j = 0; j < 4; ++j) {
    int c = tid + j * 128;           // chunk id 0..511
    int r = c >> 3, c16 = c & 7;     // row, 16B-chunk within row
    unsigned lds = (unsigned)(uintptr_t)(l + r * 72 + c16 * 8);
    const __bf16* gp = g + (size_t)r * DKV + c16 * 8;
    asm volatile("global_load_async_to_lds_b128 %0, %1, off"
                 :: "v"(lds), "v"(gp) : "memory");
  }
}

// ---------------------------------------------------------------------------
// Kernel 1: fused QKV projection.  X[32768x512] @ W[512x64] + b -> bf16 ws.
// Memory-bound (192 MiB fp32 reads); bf16 WMMA with fp32 accumulate.
// ---------------------------------------------------------------------------
__global__ __launch_bounds__(128)
void proj_kernel(const float* __restrict__ q, const float* __restrict__ k,
                 const float* __restrict__ v,
                 const float* __restrict__ Wq, const float* __restrict__ Wk,
                 const float* __restrict__ Wv,
                 const float* __restrict__ bq, const float* __restrict__ bk,
                 const float* __restrict__ bv,
                 __bf16* __restrict__ ws) {
  __shared__ alignas(16) __bf16 sX[64 * 136];   // [row][k]  padded
  __shared__ alignas(16) __bf16 sWT[64 * 136];  // [n][k]    W transposed

  const int tid  = threadIdx.x;
  const int wave = tid >> 5;
  const int lane = tid & 31;
  const int n    = lane & 15;
  const int half = lane >> 4;

  const int mi = blockIdx.y;
  const float* X    = (mi == 0) ? q  : (mi == 1) ? k  : v;
  const float* W    = (mi == 0) ? Wq : (mi == 1) ? Wk : Wv;
  const float* bias = (mi == 0) ? bq : (mi == 1) ? bk : bv;
  __bf16* out = ws + (size_t)mi * ((size_t)BATCH * SLEN * DKV);

  const int rows0 = blockIdx.x * 64;
  const int m0    = wave * 16;

  const v8f vzero = {0.f, 0.f, 0.f, 0.f, 0.f, 0.f, 0.f, 0.f};
  v8f acc[4];
#pragma unroll
  for (int t = 0; t < 4; ++t) acc[t] = vzero;

  for (int kc = 0; kc < DM; kc += 128) {
    __syncthreads();
    for (int i = tid; i < 64 * 128; i += 128) {
      int r = i >> 7, c = i & 127;
      sX[r * 136 + c] = (__bf16)X[(size_t)(rows0 + r) * DM + kc + c];
    }
    for (int i = tid; i < 128 * 64; i += 128) {
      int kl = i >> 6, nn = i & 63;
      sWT[nn * 136 + kl] = (__bf16)W[(size_t)(kc + kl) * DKV + nn];
    }
    __syncthreads();
#pragma unroll
    for (int ks = 0; ks < 128; ks += 32) {
      v16bf a = load_frag(&sX[(m0 + n) * 136 + ks + 8 * half],
                          &sX[(m0 + n) * 136 + ks + 16 + 8 * half]);
      v16bf bf[4];
#pragma unroll
      for (int nt = 0; nt < 4; ++nt)
        bf[nt] = load_frag(&sWT[(nt * 16 + n) * 136 + ks + 8 * half],
                           &sWT[(nt * 16 + n) * 136 + ks + 16 + 8 * half]);
#pragma unroll
      for (int nt = 0; nt < 4; ++nt) acc[nt] = wmma_bf16(a, bf[nt], acc[nt]);
    }
  }

#pragma unroll
  for (int nt = 0; nt < 4; ++nt) {
    float bval = bias[nt * 16 + n];
#pragma unroll
    for (int vv = 0; vv < 8; ++vv) {
      int row = rows0 + m0 + vv + 8 * half;
      out[(size_t)row * DKV + nt * 16 + n] = (__bf16)(acc[nt][vv] + bval);
    }
  }
}

// ---------------------------------------------------------------------------
// Kernel 2: flash attention, bf16 WMMA, online softmax, key-padding mask.
// K tiles double-buffered via global_load_async_to_lds_b128 (ASYNCcnt) so the
// next chunk's transfer overlaps the current chunk's 16 WMMAs + softmax.
// ---------------------------------------------------------------------------
__global__ __launch_bounds__(128)
void attn_kernel(const __bf16* __restrict__ ws, const int* __restrict__ mask,
                 float* __restrict__ out) {
  __shared__ alignas(16) __bf16 sQ[64 * 72];        // [qrow][d]
  __shared__ alignas(16) __bf16 sK[2][64 * 72];     // [key][d], double buffered
  __shared__ alignas(16) __bf16 sVT[2][64 * 72];    // [dv][key], double buffered
  __shared__ alignas(16) __bf16 sP[4 * 16 * 72];    // per-wave P tile

  const int tid  = threadIdx.x;
  const int wave = tid >> 5;
  const int lane = tid & 31;
  const int n    = lane & 15;
  const int half = lane >> 4;
  const int m0   = wave * 16;

  const int b  = blockIdx.y;
  const int q0 = blockIdx.x * 64;

  const size_t NELT = (size_t)BATCH * SLEN * DKV;
  const __bf16* qb = ws + ((size_t)b * SLEN + q0) * DKV;
  const __bf16* kb = ws + NELT + (size_t)b * SLEN * DKV;
  const __bf16* vb = ws + 2 * NELT + (size_t)b * SLEN * DKV;
  const int*    mb = mask + b * SLEN;

  // prologue: async-stage Q and K chunk 0; VGPR-stage (transpose) V chunk 0
  async_tile_64x64(qb, sQ, tid);
  async_tile_64x64(kb, sK[0], tid);
  for (int i = tid; i < 64 * 64; i += 128) {
    int r = i >> 6, c = i & 63;
    sVT[0][c * 72 + r] = vb[(size_t)r * DKV + c];
  }

  const v8f vzero = {0.f, 0.f, 0.f, 0.f, 0.f, 0.f, 0.f, 0.f};
  v8f o[4];
#pragma unroll
  for (int t = 0; t < 4; ++t) o[t] = vzero;
  float mrow[8], lrow[8];
#pragma unroll
  for (int t = 0; t < 8; ++t) { mrow[t] = -3.402823e38f; lrow[t] = 0.f; }

  const float scale = 0.125f;  // 1/sqrt(d_k=64)

  for (int it = 0; it < SLEN / 64; ++it) {
    const int s0  = it * 64;
    const int cur = it & 1;
    const int nxt = cur ^ 1;

    // publish previous async writes, then barrier: buffers [cur] now valid
    asm volatile("s_wait_asynccnt 0x0" ::: "memory");
    __syncthreads();

    // kick off next chunk's staging; overlaps with compute below
    if (it + 1 < SLEN / 64) {
      async_tile_64x64(kb + (size_t)(s0 + 64) * DKV, sK[nxt], tid);
      for (int i = tid; i < 64 * 64; i += 128) {
        int r = i >> 6, c = i & 63;
        sVT[nxt][c * 72 + r] = vb[(size_t)(s0 + 64 + r) * DKV + c];
      }
    }

    // S = Q Kᵀ : M=16 q rows, N=64 keys, K=64 d
    v8f sc[4];
#pragma unroll
    for (int t = 0; t < 4; ++t) sc[t] = vzero;
#pragma unroll
    for (int ks = 0; ks < 64; ks += 32) {
      v16bf a = load_frag(&sQ[(m0 + n) * 72 + ks + 8 * half],
                          &sQ[(m0 + n) * 72 + ks + 16 + 8 * half]);
      v16bf bf[4];
#pragma unroll
      for (int nt = 0; nt < 4; ++nt)
        bf[nt] = load_frag(&sK[cur][(nt * 16 + n) * 72 + ks + 8 * half],
                           &sK[cur][(nt * 16 + n) * 72 + ks + 16 + 8 * half]);
#pragma unroll
      for (int nt = 0; nt < 4; ++nt) sc[nt] = wmma_bf16(a, bf[nt], sc[nt]);
    }

    // scale + key-padding mask
#pragma unroll
    for (int nt = 0; nt < 4; ++nt) {
      float madd = (mb[s0 + nt * 16 + n] == 0) ? -1e10f : 0.f;
#pragma unroll
      for (int vv = 0; vv < 8; ++vv) sc[nt][vv] = sc[nt][vv] * scale + madd;
    }

    // online softmax (rows live in 16-lane halves of the C layout)
    float mnew[8], corr[8];
#pragma unroll
    for (int vv = 0; vv < 8; ++vv) {
      float t = fmaxf(fmaxf(sc[0][vv], sc[1][vv]), fmaxf(sc[2][vv], sc[3][vv]));
#pragma unroll
      for (int off = 1; off < 16; off <<= 1) t = fmaxf(t, __shfl_xor(t, off, 32));
      mnew[vv] = fmaxf(mrow[vv], t);
      corr[vv] = __expf(mrow[vv] - mnew[vv]);
      mrow[vv] = mnew[vv];
    }
#pragma unroll
    for (int nt = 0; nt < 4; ++nt)
#pragma unroll
      for (int vv = 0; vv < 8; ++vv)
        sc[nt][vv] = __expf(sc[nt][vv] - mnew[vv]);
#pragma unroll
    for (int vv = 0; vv < 8; ++vv) {
      float rs = sc[0][vv] + sc[1][vv] + sc[2][vv] + sc[3][vv];
#pragma unroll
      for (int off = 1; off < 16; off <<= 1) rs += __shfl_xor(rs, off, 32);
      lrow[vv] = lrow[vv] * corr[vv] + rs;
    }
#pragma unroll
    for (int nt = 0; nt < 4; ++nt)
#pragma unroll
      for (int vv = 0; vv < 8; ++vv) o[nt][vv] *= corr[vv];

    // P (C layout) -> LDS -> A layout, wave-private tile
    __bf16* sPw = &sP[wave * 16 * 72];
#pragma unroll
    for (int nt = 0; nt < 4; ++nt)
#pragma unroll
      for (int vv = 0; vv < 8; ++vv)
        sPw[(vv + 8 * half) * 72 + nt * 16 + n] = (__bf16)sc[nt][vv];

    // O += P V : M=16 q rows, N=64 dv, K=64 keys
#pragma unroll
    for (int ks = 0; ks < 64; ks += 32) {
      v16bf a = load_frag(&sPw[n * 72 + ks + 8 * half],
                          &sPw[n * 72 + ks + 16 + 8 * half]);
      v16bf bf[4];
#pragma unroll
      for (int nt = 0; nt < 4; ++nt)
        bf[nt] = load_frag(&sVT[cur][(nt * 16 + n) * 72 + ks + 8 * half],
                           &sVT[cur][(nt * 16 + n) * 72 + ks + 16 + 8 * half]);
#pragma unroll
      for (int nt = 0; nt < 4; ++nt) o[nt] = wmma_bf16(a, bf[nt], o[nt]);
    }
  }

  // normalize by row sum and store fp32 output
  float* ob = out + ((size_t)b * SLEN + q0) * DKV;
#pragma unroll
  for (int vv = 0; vv < 8; ++vv) {
    float inv = 1.0f / lrow[vv];
    int row = m0 + vv + 8 * half;
#pragma unroll
    for (int nt = 0; nt < 4; ++nt)
      ob[(size_t)row * DKV + nt * 16 + n] = o[nt][vv] * inv;
  }
}

extern "C" void kernel_launch(void* const* d_in, const int* in_sizes, int n_in,
                              void* d_out, int out_size, void* d_ws, size_t ws_size,
                              hipStream_t stream) {
  const float* q   = (const float*)d_in[0];
  const float* k   = (const float*)d_in[1];
  const float* v   = (const float*)d_in[2];
  const int*   msk = (const int*)d_in[3];
  const float* Wq  = (const float*)d_in[4];
  const float* bq  = (const float*)d_in[5];
  const float* Wk  = (const float*)d_in[6];
  const float* bk  = (const float*)d_in[7];
  const float* Wv  = (const float*)d_in[8];
  const float* bv  = (const float*)d_in[9];
  float*  out = (float*)d_out;
  __bf16* ws  = (__bf16*)d_ws;  // qh | kh | vh, 4 MiB each (bf16)

  dim3 g1(BATCH * SLEN / 64, 3), b1(128);
  proj_kernel<<<g1, b1, 0, stream>>>(q, k, v, Wq, Wk, Wv, bq, bk, bv, ws);

  dim3 g2(SLEN / 64, BATCH), b2(128);
  attn_kernel<<<g2, b2, 0, stream>>>(ws, msk, out);
}
